// SIFTDescriptor_8272107012368
// MI455X (gfx1250) — compile-verified, matrix operand order checked by reading, not run on hardware
//
#include <hip/hip_runtime.h>

typedef __attribute__((ext_vector_type(16))) _Float16 v16h;
typedef __attribute__((ext_vector_type(8)))  float    v8f;

#define PATCH   41
#define NUM_ANG 8
#define NP      4                        // patches per workgroup
#define KROW    72                       // padded row stride in halves (36 dwords, bank-friendly)
#define NROWS   (NP * NUM_ANG * 4)       // 128 (patch, angle, ox) rows

union HPack { v16h h; unsigned int u[8]; };

__global__ __launch_bounds__(256)
void sift_desc_kernel(const float* __restrict__ x,
                      const float* __restrict__ gk,
                      float* __restrict__ out, int nB)
{
    __shared__ unsigned int pxu[NROWS * KROW / 2];   // 18.4 KB, dword-aligned f16 rows
    __shared__ float        descb[NP * 128];         // 2 KB raw descriptors
    _Float16* pxh = (_Float16*)pxu;

    const int tid  = threadIdx.x;
    const int lane = tid & 31;
    const int wave = tid >> 5;

    // ---- zero px tile (covers K padding; A rows are zero beyond K=40) ----
    #pragma unroll
    for (int i = 0; i < (NROWS * KROW / 2) / 256; ++i)
        pxu[tid + 256 * i] = 0u;
    __syncthreads();

    // ---- Phase 1: per-(patch,row) gradients, soft angular bins, X-pooling ----
    if (tid < NP * PATCH) {
        const int pl = tid / PATCH;
        const int y  = tid % PATCH;
        const int pg = blockIdx.x * NP + pl;
        if (pg < nB) {
            const float* base = x + (size_t)pg * (PATCH * PATCH);
            const int ym = (y > 0) ? y - 1 : 0;
            const int yp = (y < PATCH - 1) ? y + 1 : PATCH - 1;
            const float* rc = base + y  * PATCH;
            const float* ru = base + ym * PATCH;
            const float* rd = base + yp * PATCH;
            const float* gr = gk + y * PATCH;

            float acc[32];
            #pragma unroll
            for (int i = 0; i < 32; ++i) acc[i] = 0.0f;

            float prv = rc[0], cur = rc[0];
            for (int xx = 0; xx < PATCH; ++xx) {
                float nxt = rc[(xx < PATCH - 1) ? xx + 1 : PATCH - 1];
                float gxv = 0.5f * (nxt - prv);
                float gyv = 0.5f * (rd[xx] - ru[xx]);
                prv = cur; cur = nxt;

                // scale by 512: keeps f16 out of subnormals; cancels in the L2 normalize
                float mag = sqrtf(gxv * gxv + gyv * gyv + 1e-10f) * gr[xx] * 512.0f;
                float ori = atan2f(gyv, gxv + 1e-10f) + 6.2831853071795864f;
                float ob  = ori * (8.0f / 6.2831853071795864f);
                float bf  = floorf(ob);
                float w1  = ob - bf;
                int   b0  = ((int)bf) & 7;
                int   b1  = (b0 + 1) & 7;
                float wo0 = (1.0f - w1) * mag;
                float wo1 = w1 * mag;

                // branch-free triangular X weights (zero outside window by construction)
                float wx[4];
                #pragma unroll
                for (int ox = 0; ox < 4; ++ox)
                    wx[ox] = fmaxf(0.0f, 8.0f - fabsf((float)xx - 3.5f - 10.0f * ox)) * 0.125f;

                // statically-indexed accumulators; bin select via cndmask, no scratch
                #pragma unroll
                for (int a = 0; a < NUM_ANG; ++a) {
                    float wa = (a == b0) ? wo0 : ((a == b1) ? wo1 : 0.0f);
                    #pragma unroll
                    for (int ox = 0; ox < 4; ++ox)
                        acc[a * 4 + ox] += wa * wx[ox];
                }
            }
            #pragma unroll
            for (int i = 0; i < 32; ++i) {
                const int m = pl * 32 + i;
                pxh[m * KROW + y] = (_Float16)acc[i];
            }
        }
    }
    __syncthreads();

    // ---- Phase 2: Y-pooling as GEMM via WMMA.
    // desc[m][oy] = sum_y px[m][y] * wy[y][oy]; 128 rows = 8 tiles, wave w owns tile w.
    // K = 64 = 2 x v_wmma_f32_16x16x32_f16. EXEC uniform throughout.
    {
        const int n     = lane & 15;
        const int khalf = lane >> 4;
        const int m     = wave * 16 + n;
        const float nf  = (float)n;

        v8f c = {};
        #pragma unroll
        for (int s = 0; s < 2; ++s) {
            HPack ap;
            v16h  b;
            #pragma unroll
            for (int v = 0; v < 8; ++v) {                 // packed pair loads: ds_load_b32
                const int K0 = s * 32 + ((v & 3) * 2) + ((v >> 2) * 16) + khalf * 8;
                ap.u[v] = *(const unsigned int*)(pxh + m * KROW + K0);
            }
            #pragma unroll
            for (int i = 0; i < 16; ++i) {                // branch-free wy weights
                const int v2 = i >> 1, j = i & 1;
                const int K  = s * 32 + ((v2 & 3) * 2 + j) + ((v2 >> 2) * 16) + khalf * 8;
                float wv = fmaxf(0.0f, 8.0f - fabsf((float)K - 3.5f - 10.0f * nf)) * 0.125f;
                b[i] = (_Float16)wv;
            }
            c = __builtin_amdgcn_wmma_f32_16x16x32_f16(
                    false, ap.h, false, b, (short)0, c, false, false);
        }
        if (n < 4) {                                      // only oy columns 0..3 are real
            #pragma unroll
            for (int v = 0; v < 8; ++v) {
                const int ml = wave * 16 + v + 8 * khalf;
                const int p  = ml >> 5;
                const int rr = ml & 31;
                const int a8 = rr >> 2;
                const int ox = rr & 3;
                descb[p * 128 + a8 * 16 + n * 4 + ox] = c[v];
            }
        }
    }
    __syncthreads();

    // ---- Phase 3: L2 -> clip 0.2 -> L2 -> RootSIFT; one wave per patch ----
    if (wave < NP) {
        const int pg = blockIdx.x * NP + wave;
        if (pg < nB) {
            float d[4];
            #pragma unroll
            for (int j = 0; j < 4; ++j) d[j] = descb[wave * 128 + j * 32 + lane];

            float s = d[0]*d[0] + d[1]*d[1] + d[2]*d[2] + d[3]*d[3];
            for (int o = 16; o > 0; o >>= 1) s += __shfl_xor(s, o, 32);
            float inv = 1.0f / fmaxf(sqrtf(s), 1e-12f);
            #pragma unroll
            for (int j = 0; j < 4; ++j)
                d[j] = fminf(fmaxf(d[j] * inv, 0.0f), 0.2f);

            s = d[0]*d[0] + d[1]*d[1] + d[2]*d[2] + d[3]*d[3];
            for (int o = 16; o > 0; o >>= 1) s += __shfl_xor(s, o, 32);
            inv = 1.0f / fmaxf(sqrtf(s), 1e-12f);
            #pragma unroll
            for (int j = 0; j < 4; ++j) d[j] *= inv;

            s = d[0] + d[1] + d[2] + d[3];
            for (int o = 16; o > 0; o >>= 1) s += __shfl_xor(s, o, 32);
            inv = 1.0f / fmaxf(s, 1e-12f);

            float* op = out + (size_t)pg * 128;
            #pragma unroll
            for (int j = 0; j < 4; ++j)
                op[j * 32 + lane] = sqrtf(d[j] * inv + 1e-10f);
        }
    }
}

extern "C" void kernel_launch(void* const* d_in, const int* in_sizes, int n_in,
                              void* d_out, int out_size, void* d_ws, size_t ws_size,
                              hipStream_t stream) {
    const float* x  = (const float*)d_in[0];   // (B,1,41,41) f32
    const float* gk = (const float*)d_in[1];   // (41,41) f32 gaussian
    // d_in[2] (16x16 pooling kernel) is separable triangular; recomputed analytically
    float* out = (float*)d_out;                // (B,128) f32

    const int nB   = in_sizes[0] / (PATCH * PATCH);
    const int grid = (nB + NP - 1) / NP;
    sift_desc_kernel<<<grid, 256, 0, stream>>>(x, gk, out, nB);
}
